// SpatioTemporalAttentionHighMem_71914932404732
// MI455X (gfx1250) — matrix operand order repaired and checked by reading, not compiled, not run on hardware
//
#include <hip/hip_runtime.h>
#include <hip/hip_bf16.h>

#define T_    3
#define H_    96
#define W_    96
#define C_    64
#define HW_   (H_ * W_)      // 9216
#define ROWS_ (T_ * HW_)     // 27648
#define NH_   4
#define KD_   16
#define P_    25
#define S_    75

typedef __attribute__((ext_vector_type(16))) __bf16 v16bf;
typedef __attribute__((ext_vector_type(8)))  __bf16 v8bf;
typedef __attribute__((ext_vector_type(8)))  float  v8f;

static __device__ inline v16bf cat8(v8bf lo, v8bf hi) {
  return __builtin_shufflevector(lo, hi, 0, 1, 2, 3, 4, 5, 6, 7,
                                         8, 9, 10, 11, 12, 13, 14, 15);
}

// ---------------------------------------------------------------------------
// SEK[p,nk] = spatial_emb[p] @ Wk + bk ; SEQ[p,nk] = spatial_emb[p] @ Wq + bq
// ---------------------------------------------------------------------------
__global__ __launch_bounds__(128)
void se_kernel(const float* __restrict__ spatial_emb,
               const float* __restrict__ Wk, const float* __restrict__ bk,
               const float* __restrict__ Wq, const float* __restrict__ bq,
               float* __restrict__ SEK, float* __restrict__ SEQ) {
  int idx = blockIdx.x * 128 + threadIdx.x;
  if (idx >= P_ * C_) return;
  int p = idx >> 6, col = idx & 63;
  float ak = bk[col], aq = bq[col];
  #pragma unroll 4
  for (int c = 0; c < C_; ++c) {
    float e = spatial_emb[p * C_ + c];
    ak += e * Wk[c * 64 + col];
    aq += e * Wq[c * 64 + col];
  }
  SEK[idx] = ak;
  SEQ[idx] = aq;
}

// ---------------------------------------------------------------------------
// Fused projection GEMM: one 16-row M tile per block, 4 waves = 4 N tiles.
// Kpre = (v+te)@Wk ; Vpre = v@Wv + bv ; Qpre = (v+te)@Wq
// WMMA bf16 16x16x32, fp32 accumulate.
// ---------------------------------------------------------------------------
__global__ __launch_bounds__(128)
void proj_kernel(const float* __restrict__ values,
                 const float* __restrict__ temp_emb,
                 const float* __restrict__ Wk, const float* __restrict__ Wv,
                 const float* __restrict__ Wq, const float* __restrict__ bv,
                 float* __restrict__ Kpre, float* __restrict__ Vpre,
                 float* __restrict__ Qpre) {
  __shared__ __bf16 Akq[16][64];   // values + temp_emb (bf16)
  __shared__ __bf16 Av [16][64];   // values
  __shared__ __bf16 Btk[64][64];   // Wk transposed: Bt[n][c]
  __shared__ __bf16 Btv[64][64];
  __shared__ __bf16 Btq[64][64];

  const int tid  = threadIdx.x;
  const int row0 = blockIdx.x * 16;

  { // A tiles: 16x64 f32, 8 contiguous elements per thread
    int base = tid * 8;
    int r = base >> 6, c = base & 63;
    int grow = row0 + r;
    int t = grow / HW_;
    const float* vp = values   + (size_t)grow * C_ + c;
    const float* te = temp_emb + t * C_ + c;
    #pragma unroll
    for (int j = 0; j < 8; ++j) {
      float v = vp[j];
      Akq[r][c + j] = (__bf16)(v + te[j]);
      Av [r][c + j] = (__bf16)v;
    }
  }
  // B tiles: store transposed so a lane reads its K elements contiguously
  #pragma unroll
  for (int i = 0; i < 32; ++i) {
    int idx = i * 128 + tid;
    int c = idx >> 6, n = idx & 63;
    Btk[n][c] = (__bf16)Wk[idx];
    Btv[n][c] = (__bf16)Wv[idx];
    Btq[n][c] = (__bf16)Wq[idx];
  }
  __syncthreads();

  const int wave = tid >> 5, lane = tid & 31;
  const int half = lane >> 4, l15 = lane & 15;
  const int N = wave * 16 + l15;

  v8f ck = {}, cv = {}, cq = {};
  #pragma unroll
  for (int kk = 0; kk < 64; kk += 32) {
    // A layout: elem j -> K=kk+half*16+j
    v16bf akq = cat8(*(const v8bf*)&Akq[l15][kk + half * 16],
                     *(const v8bf*)&Akq[l15][kk + half * 16 + 8]);
    v16bf av_ = cat8(*(const v8bf*)&Av [l15][kk + half * 16],
                     *(const v8bf*)&Av [l15][kk + half * 16 + 8]);
    // B layout: elem j -> K = kk + half*16 + j, col = l15 (contiguous in Bt[N])
    v16bf bk_ = cat8(*(const v8bf*)&Btk[N][kk + half * 16],
                     *(const v8bf*)&Btk[N][kk + half * 16 + 8]);
    v16bf bv_ = cat8(*(const v8bf*)&Btv[N][kk + half * 16],
                     *(const v8bf*)&Btv[N][kk + half * 16 + 8]);
    v16bf bq_ = cat8(*(const v8bf*)&Btq[N][kk + half * 16],
                     *(const v8bf*)&Btq[N][kk + half * 16 + 8]);
    ck = __builtin_amdgcn_wmma_f32_16x16x32_bf16(false, akq, false, bk_, (short)0, ck, false, false);
    cv = __builtin_amdgcn_wmma_f32_16x16x32_bf16(false, av_, false, bv_, (short)0, cv, false, false);
    cq = __builtin_amdgcn_wmma_f32_16x16x32_bf16(false, akq, false, bq_, (short)0, cq, false, false);
  }

  const float bvn = bv[N];
  #pragma unroll
  for (int r = 0; r < 8; ++r) {
    int m = half * 8 + r;                 // C/D layout: M = half*8 + r, N = l15
    size_t o = (size_t)(row0 + m) * C_ + N;
    Kpre[o] = ck[r];
    Vpre[o] = cv[r] + bvn;
    Qpre[o] = cq[r];
  }
}

// ---------------------------------------------------------------------------
// Attention: one wave32 per pixel; lanes own s = lane, lane+32, lane+64.
// All 3 query time-steps share a single sweep over the 75 keys/values
// (3x less K/V traffic); SEK/SEQ staged in LDS once per block.
// ---------------------------------------------------------------------------
__global__ __launch_bounds__(256)
void attn_kernel(const float* __restrict__ Kpre, const float* __restrict__ Vpre,
                 const float* __restrict__ Qpre, const float* __restrict__ SEKg,
                 const float* __restrict__ SEQg, float* __restrict__ Ctx) {
  __shared__ float sek[P_ * C_];   // 6.4 KB
  __shared__ float seq[P_ * C_];   // 6.4 KB
  const int tid = threadIdx.x;
  for (int i = tid; i < P_ * C_; i += 256) {
    sek[i] = SEKg[i];
    seq[i] = SEQg[i];
  }
  __syncthreads();

  const int lane = tid & 31;
  const int pix  = blockIdx.x * 8 + (tid >> 5);
  if (pix >= HW_) return;
  const int h = pix / W_, w = pix % W_;
  const int cy = min(max(h, 2), H_ - 3);
  const int cx = min(max(w, 2), W_ - 3);
  const int qp = (h - cy + 2) * 5 + (w - cx + 2);

  int  sBase[3], sSE[3];
  bool sv[3];
  #pragma unroll
  for (int i = 0; i < 3; ++i) {
    int s = lane + 32 * i;
    sv[i] = (s < S_);
    int ss = sv[i] ? s : 0;
    int t2 = ss / P_, p = ss % P_;
    int py = cy + p / 5 - 2, px = cx + p % 5 - 2;
    sBase[i] = ((t2 * HW_) + py * W_ + px) * C_;
    sSE[i]   = p * C_;
  }

  const float scale = 0.25f;  // 1/sqrt(KD)
  #pragma unroll
  for (int n = 0; n < NH_; ++n) {
    const int hoff = n * KD_;
    // All three query vectors for this head
    float q[3][16];
    #pragma unroll
    for (int t = 0; t < T_; ++t) {
      const float4* qa = (const float4*)(Qpre + (size_t)(t * HW_ + pix) * C_ + hoff);
      const float4* qs = (const float4*)(seq + qp * C_ + hoff);
      #pragma unroll
      for (int j = 0; j < 4; ++j) {
        float4 a = qa[j], b = qs[j];
        q[t][4*j+0] = a.x + b.x; q[t][4*j+1] = a.y + b.y;
        q[t][4*j+2] = a.z + b.z; q[t][4*j+3] = a.w + b.w;
      }
    }
    // One sweep over keys: scores for all 3 queries
    float sc[3][3];
    float mx[3] = {-3.0e38f, -3.0e38f, -3.0e38f};
    #pragma unroll
    for (int i = 0; i < 3; ++i) {
      #pragma unroll
      for (int t = 0; t < T_; ++t) sc[t][i] = -3.0e38f;
      if (sv[i]) {
        const float4* kr = (const float4*)(Kpre + sBase[i] + hoff);
        const float4* se = (const float4*)(sek + sSE[i] + hoff);
        float kv[16];
        #pragma unroll
        for (int j = 0; j < 4; ++j) {
          float4 a = kr[j], b = se[j];
          kv[4*j+0] = a.x + b.x; kv[4*j+1] = a.y + b.y;
          kv[4*j+2] = a.z + b.z; kv[4*j+3] = a.w + b.w;
        }
        #pragma unroll
        for (int t = 0; t < T_; ++t) {
          float acc = 0.f;
          #pragma unroll
          for (int k = 0; k < 16; ++k) acc += q[t][k] * kv[k];
          sc[t][i] = acc * scale;
          mx[t] = fmaxf(mx[t], sc[t][i]);
        }
      }
    }
    #pragma unroll
    for (int off = 16; off > 0; off >>= 1)
      #pragma unroll
      for (int t = 0; t < T_; ++t) mx[t] = fmaxf(mx[t], __shfl_xor(mx[t], off, 32));

    float e[3][3], sum[3] = {0.f, 0.f, 0.f};
    #pragma unroll
    for (int t = 0; t < T_; ++t)
      #pragma unroll
      for (int i = 0; i < 3; ++i) {
        e[t][i] = sv[i] ? __expf(sc[t][i] - mx[t]) : 0.f;
        sum[t] += e[t][i];
      }
    #pragma unroll
    for (int off = 16; off > 0; off >>= 1)
      #pragma unroll
      for (int t = 0; t < T_; ++t) sum[t] += __shfl_xor(sum[t], off, 32);
    float inv[3];
    #pragma unroll
    for (int t = 0; t < T_; ++t) inv[t] = 1.0f / sum[t];

    // One sweep over values: ctx for all 3 queries
    float c3[3][16];
    #pragma unroll
    for (int t = 0; t < T_; ++t)
      #pragma unroll
      for (int k = 0; k < 16; ++k) c3[t][k] = 0.f;
    #pragma unroll
    for (int i = 0; i < 3; ++i) {
      if (sv[i]) {
        const float4* vr = (const float4*)(Vpre + sBase[i] + hoff);
        float vv[16];
        #pragma unroll
        for (int j = 0; j < 4; ++j) {
          float4 a = vr[j];
          vv[4*j+0] = a.x; vv[4*j+1] = a.y; vv[4*j+2] = a.z; vv[4*j+3] = a.w;
        }
        #pragma unroll
        for (int t = 0; t < T_; ++t) {
          float wg = e[t][i] * inv[t];
          #pragma unroll
          for (int k = 0; k < 16; ++k) c3[t][k] += wg * vv[k];
        }
      }
    }
    #pragma unroll
    for (int off = 16; off > 0; off >>= 1)
      #pragma unroll
      for (int t = 0; t < T_; ++t)
        #pragma unroll
        for (int k = 0; k < 16; ++k) c3[t][k] += __shfl_xor(c3[t][k], off, 32);

    #pragma unroll
    for (int t = 0; t < T_; ++t) {
      float outv = 0.f;
      #pragma unroll
      for (int k = 0; k < 16; ++k) outv = (lane == k) ? c3[t][k] : outv;
      if (lane < 16)
        Ctx[(size_t)(t * HW_ + pix) * C_ + hoff + lane] = outv;
    }
  }
}

// ---------------------------------------------------------------------------
// Output GEMM: out[pix, t*64+c] = Ctx[t*HW+pix, :] @ Wo[:, c] + bo[c]
// ---------------------------------------------------------------------------
__global__ __launch_bounds__(128)
void out_kernel(const float* __restrict__ Ctx, const float* __restrict__ Wo,
                const float* __restrict__ bo, float* __restrict__ out) {
  __shared__ __bf16 Ac [16][64];
  __shared__ __bf16 Btw[64][64];
  const int tid  = threadIdx.x;
  const int row0 = blockIdx.x * 16;
  {
    int base = tid * 8, r = base >> 6, c = base & 63;
    const float* sp = Ctx + (size_t)(row0 + r) * C_ + c;
    #pragma unroll
    for (int j = 0; j < 8; ++j) Ac[r][c + j] = (__bf16)sp[j];
  }
  #pragma unroll
  for (int i = 0; i < 32; ++i) {
    int idx = i * 128 + tid;
    int k = idx >> 6, n = idx & 63;
    Btw[n][k] = (__bf16)Wo[idx];   // Wo flat [nk=64][c=64], transpose into LDS
  }
  __syncthreads();

  const int wave = tid >> 5, lane = tid & 31;
  const int half = lane >> 4, l15 = lane & 15;
  const int N = wave * 16 + l15;
  v8f acc = {};
  #pragma unroll
  for (int kk = 0; kk < 64; kk += 32) {
    v16bf a = cat8(*(const v8bf*)&Ac [l15][kk + half * 16],
                   *(const v8bf*)&Ac [l15][kk + half * 16 + 8]);
    v16bf b = cat8(*(const v8bf*)&Btw[N][kk + half * 16],
                   *(const v8bf*)&Btw[N][kk + half * 16 + 8]);
    acc = __builtin_amdgcn_wmma_f32_16x16x32_bf16(false, a, false, b, (short)0, acc, false, false);
  }
  const float bon = bo[N];
  #pragma unroll
  for (int r = 0; r < 8; ++r) {
    int grow = row0 + half * 8 + r;
    int t = grow / HW_, pix = grow % HW_;
    out[(size_t)pix * (T_ * C_) + t * C_ + N] = acc[r] + bon;
  }
}

// ---------------------------------------------------------------------------
extern "C" void kernel_launch(void* const* d_in, const int* in_sizes, int n_in,
                              void* d_out, int out_size, void* d_ws, size_t ws_size,
                              hipStream_t stream) {
  const float* values      = (const float*)d_in[0];
  const float* temp_emb    = (const float*)d_in[1];
  const float* spatial_emb = (const float*)d_in[2];
  const float* Wq          = (const float*)d_in[3];
  const float* bq          = (const float*)d_in[4];
  const float* Wk          = (const float*)d_in[5];
  const float* bk          = (const float*)d_in[6];
  const float* Wv          = (const float*)d_in[7];
  const float* bv          = (const float*)d_in[8];
  const float* Wo          = (const float*)d_in[9];
  const float* bo          = (const float*)d_in[10];
  float* out = (float*)d_out;
  float* ws  = (float*)d_ws;

  const size_t R = (size_t)ROWS_ * C_;   // 1,769,472 floats per plane
  float* Kpre = ws;
  float* Vpre = ws + R;
  float* Qpre = ws + 2 * R;
  float* Ctx  = ws + 3 * R;
  float* SEK  = ws + 4 * R;
  float* SEQ  = SEK + P_ * C_;

  se_kernel  <<<(P_ * C_ + 127) / 128, 128, 0, stream>>>(spatial_emb, Wk, bk, Wq, bq, SEK, SEQ);
  proj_kernel<<<ROWS_ / 16, 128, 0, stream>>>(values, temp_emb, Wk, Wv, Wq, bv, Kpre, Vpre, Qpre);
  attn_kernel<<<HW_ / 8, 256, 0, stream>>>(Kpre, Vpre, Qpre, SEK, SEQ, Ctx);
  out_kernel <<<ROWS_ / 16, 128, 0, stream>>>(Ctx, Wo, bo, out);
}
